// SAModule_21809843929149
// MI455X (gfx1250) — compile-verified, matrix operand order checked by reading, not run on hardware
//
#include <hip/hip_runtime.h>
#include <type_traits>

typedef __attribute__((ext_vector_type(16))) _Float16 v16h;
typedef __attribute__((ext_vector_type(8)))  float    v8f;
typedef __attribute__((ext_vector_type(4)))  unsigned v4u;
typedef __attribute__((ext_vector_type(8)))  int      v8i;
typedef __attribute__((ext_vector_type(4)))  int      v4i;

#define IN_CH   64
#define H1G     68
#define H2C     128
#define ATTN_C  32
#define E4C     512
#define RNNC    128
#define KNN     16
#define NBATCH  8
#define KP1     96      // 68 padded to 3 K-tiles of 32
#define BN_EPS  1e-5f

// ---------------- WMMA fragment helpers (ISA 7.12.2 layouts) ----------------
// A (16x32 f16): lane l -> row l%16 ; VGPR v holds K pair at
//   K = (v>=4?16:0) + (l>=16?8:0) + (v%4)*2
// B (32x16 f16) symmetric with lane = column; weights stored transposed [N][K].
__device__ __forceinline__ v16h load_frag(const _Float16* base, int stride, int kbase) {
  int lane = threadIdx.x & 31;
  const _Float16* q = base + (size_t)(lane & 15) * stride + kbase + ((lane >> 4) << 3);
  union { v16h v; unsigned u[8]; } f;
#pragma unroll
  for (int i = 0; i < 8; ++i) {
    int off = ((i & 4) << 2) + ((i & 3) << 1); // (i>=4?16:0) + (i%4)*2
    f.u[i] = *(const unsigned*)(q + off);
  }
  return f.v;
}

template<int KT>
__device__ __forceinline__ v8f gemm16(const _Float16* A, int astride,
                                      const _Float16* BT, int kdim, int ncolbase) {
  v8f acc = {};
  const _Float16* Bb = BT + (size_t)ncolbase * kdim;
#pragma unroll
  for (int kt = 0; kt < KT; ++kt) {
    v16h a = load_frag(A,  astride, kt * 32);
    v16h b = load_frag(Bb, kdim,    kt * 32);
    acc = __builtin_amdgcn_wmma_f32_16x16x32_f16(false, a, false, b, (short)0, acc, false, false);
  }
  return acc;
}

// bn(z,p) = s*z + t with s = p0/sqrt(p3+eps), t = p1 - p2*s  (p is (4,C))
__device__ __forceinline__ void bn_coef(const float* bn, int C, int c, float& s, float& t) {
  float g = bn[c], b = bn[C + c], mu = bn[2 * C + c], var = bn[3 * C + c];
  s = g * __frsqrt_rn(var + BN_EPS);
  t = b - mu * s;
}

// ---------------- Tensor Data Mover: 1-D LDS panel stage ----------------
// D# group0: count=1 (user), lds_addr[63:32], global_addr[120:64], type=2.
// D# group1: data_size=2B, tensor_dim0 = tile_dim0 = nelem halves, dims1.. = 1.
__device__ __forceinline__ unsigned lds_addr_of(const void* p) {
  return (unsigned)(unsigned long long)p;
}
__device__ __forceinline__ void tdm_stage_1d(unsigned ldsa, const _Float16* src, unsigned nelem) {
  unsigned long long ga = (unsigned long long)(const void*)src;
  v4u g0;
  g0.x = 1u;                                        // count=1
  g0.y = ldsa;                                      // LDS byte address
  g0.z = (unsigned)(ga & 0xFFFFFFFFu);              // global addr lo
  g0.w = (unsigned)((ga >> 32) & 0x01FFFFFFu) | (2u << 30); // addr hi | type=2
  v8i g1;
  g1[0] = 1 << 16;                                  // workgroup_mask=0, data_size=1 (2B)
  g1[1] = (int)((nelem & 0xFFFFu) << 16);           // tensor_dim0 lo16 @ bits[63:48]
  g1[2] = (int)(((nelem >> 16) & 0xFFFFu) | (1u << 16)); // dim0 hi16 | tensor_dim1=1
  g1[3] = (int)((nelem & 0xFFFFu) << 16);           // tile_dim0 @ bits[127:112]
  g1[4] = 1;                                        // tile_dim1=1, tile_dim2=0
  g1[5] = (int)nelem;                               // tensor_dim0_stride lo32
  g1[6] = 0;
  g1[7] = 0;
  v4i z4 = {0, 0, 0, 0};
  v8i z8 = {0, 0, 0, 0, 0, 0, 0, 0};
  // amdgpu-toolchain (clang-23) 6-arg form: (g0, g1, g2, g3, g4, cpol)
  __builtin_amdgcn_tensor_load_to_lds(g0, g1, z4, z4, z8, 0);
}

// ---------------- small utility kernels ----------------
__global__ void zero_u32_kernel(unsigned* p, int n) {
  int i = blockIdx.x * blockDim.x + threadIdx.x;
  if (i < n) p[i] = 0u;
}

// transpose f32 [K][N] -> f16 [N][Kpad] (zero pad K)
__global__ void cvtT_kernel(const float* __restrict__ src, _Float16* __restrict__ dst,
                            int Kdim, int Ndim, int Kpad) {
  int i = blockIdx.x * blockDim.x + threadIdx.x;
  if (i >= Ndim * Kpad) return;
  int n = i / Kpad, k = i - n * Kpad;
  dst[i] = (k < Kdim) ? (_Float16)src[(size_t)k * Ndim + n] : (_Float16)0.f;
}

// ---------------- phase 1: gating MLP + segment max ----------------
__global__ void gate_mlp_kernel(const float* __restrict__ pos, const float* __restrict__ refl,
                                const int* __restrict__ batch,
                                const float* __restrict__ w1, const float* __restrict__ b1,
                                const float* __restrict__ w2, const float* __restrict__ b2,
                                unsigned* __restrict__ pooledBits, int N) {
  int n = blockIdx.x * blockDim.x + threadIdx.x;
  if (n >= N) return;
  float f0 = pos[n * 3], f1 = pos[n * 3 + 1], f2 = pos[n * 3 + 2], f3 = refl[n];
  float h[H1G];
#pragma unroll 4
  for (int j = 0; j < H1G; ++j) {
    float v = f0 * w1[j] + f1 * w1[H1G + j] + f2 * w1[2 * H1G + j] + f3 * w1[3 * H1G + j] + b1[j];
    h[j] = v > 0.f ? v : 0.f;
  }
  int b = batch[n];
  for (int j = 0; j < H1G; ++j) {
    float v = b2[j];
#pragma unroll 4
    for (int c = 0; c < H1G; ++c) v += h[c] * w2[c * H1G + j];
    v = v > 0.f ? v : 0.f;                       // relu -> >=0, uint order == float order
    atomicMax(&pooledBits[b * H1G + j], __float_as_uint(v));
  }
}

__global__ void gate_softmax_kernel(const unsigned* __restrict__ pooledBits,
                                    const float* __restrict__ wg, const float* __restrict__ bg,
                                    const float* __restrict__ gum, float* __restrict__ gsel) {
  int b = threadIdx.x;
  if (b >= NBATCH) return;
  float l0 = bg[0], l1 = bg[1];
  for (int c = 0; c < H1G; ++c) {
    float p = __uint_as_float(pooledBits[b * H1G + c]);
    l0 += p * wg[c * 2]; l1 += p * wg[c * 2 + 1];
  }
  l0 += gum[b * 2]; l1 += gum[b * 2 + 1];
  float mx = fmaxf(l0, l1);
  float e0 = __expf(l0 - mx), e1 = __expf(l1 - mx);
  gsel[b] = e1 / (e0 + e1);
}

__global__ void pos4_kernel(const float* __restrict__ pos, const float* __restrict__ refl,
                            const int* __restrict__ batch, const float* __restrict__ gsel,
                            float* __restrict__ pos4, int N) {
  int n = blockIdx.x * blockDim.x + threadIdx.x;
  if (n >= N) return;
  pos4[n * 4 + 0] = pos[n * 3 + 0];
  pos4[n * 4 + 1] = pos[n * 3 + 1];
  pos4[n * 4 + 2] = pos[n * 3 + 2];
  pos4[n * 4 + 3] = gsel[batch[n]] * refl[n];
}

// ---------------- phase 2: edge message passing + attention (WMMA) ----------------
// One wave handles one center: its 16 KNN edges are the 16 M-rows of WMMA tiles.
#define P2_WAVES 4
__global__ void __launch_bounds__(P2_WAVES * 32)
edge_attn_kernel(const float* __restrict__ x, const float* __restrict__ pos4,
                 const int* __restrict__ idx, const int* __restrict__ esrc,
                 const _Float16* __restrict__ w1T, const float* __restrict__ lb1,
                 const _Float16* __restrict__ w2T, const float* __restrict__ lb2,
                 const _Float16* __restrict__ aT,  const float* __restrict__ ab1,
                 const float* __restrict__ aw2,
                 float* __restrict__ agg, _Float16* __restrict__ aggh, int M) {
  __shared__ _Float16 sIn [P2_WAVES][16][KP1];
  __shared__ _Float16 sHm [P2_WAVES][16][H2C];
  __shared__ _Float16 sMsg[P2_WAVES][16][H2C];
  __shared__ float    sScore[P2_WAVES][16];

  int tid = threadIdx.x, wave = tid >> 5, lane = tid & 31;
  int col = lane & 15, rbase = (lane >> 4) << 3;
  int tiles = (M + P2_WAVES - 1) / P2_WAVES;

  for (int t = blockIdx.x; t < tiles; t += gridDim.x) {
    int m = t * P2_WAVES + wave;
    if (m >= M) continue;
    // prefetch next tile's edge list (global_prefetch_b8)
    __builtin_prefetch(esrc + (size_t)(t + gridDim.x) * P2_WAVES * KNN, 0, 3);
    int cpt = idx[m];
    float c0 = pos4[cpt * 4 + 0], c1 = pos4[cpt * 4 + 1];
    float c2 = pos4[cpt * 4 + 2], c3 = pos4[cpt * 4 + 3];
    const int* sp = esrc + (size_t)m * KNN;

    // assemble 16 x 96 input tile: [x[src] (64) | pos4[src]-pos4[center] (4) | 0 pad]
    for (int j = lane; j < 16 * KP1; j += 32) {
      int e = j / KP1, c = j - e * KP1;
      float v = 0.f;
      int s = sp[e];
      if (c < IN_CH) v = x[(size_t)s * IN_CH + c];
      else if (c < IN_CH + 4) {
        int d = c - IN_CH;
        float ce = (d == 0) ? c0 : (d == 1) ? c1 : (d == 2) ? c2 : c3;
        v = pos4[s * 4 + d] - ce;
      }
      sIn[wave][e][c] = (_Float16)v;
    }
    if (lane < 16) sScore[wave][lane] = 0.f;

    // GEMM1: (16x96) @ (96x128) -> relu -> hm
#pragma unroll
    for (int nt = 0; nt < H2C / 16; ++nt) {
      v8f acc = gemm16<3>(&sIn[wave][0][0], KP1, w1T, KP1, nt * 16);
      float bias = lb1[nt * 16 + col];
#pragma unroll
      for (int i = 0; i < 8; ++i) {
        float v = acc[i] + bias;
        sHm[wave][rbase + i][nt * 16 + col] = (_Float16)(v > 0.f ? v : 0.f);
      }
    }
    // GEMM2: (16x128) @ (128x128) -> relu -> msg
#pragma unroll
    for (int nt = 0; nt < H2C / 16; ++nt) {
      v8f acc = gemm16<4>(&sHm[wave][0][0], H2C, w2T, H2C, nt * 16);
      float bias = lb2[nt * 16 + col];
#pragma unroll
      for (int i = 0; i < 8; ++i) {
        float v = acc[i] + bias;
        sMsg[wave][rbase + i][nt * 16 + col] = (_Float16)(v > 0.f ? v : 0.f);
      }
    }
    // attention scores: tanh(msg @ attn_w1 + b) @ attn_w2
#pragma unroll
    for (int nt = 0; nt < ATTN_C / 16; ++nt) {
      v8f acc = gemm16<4>(&sMsg[wave][0][0], H2C, aT, H2C, nt * 16);
      int c = nt * 16 + col;
      float wb = ab1[c], wv = aw2[c];
#pragma unroll
      for (int i = 0; i < 8; ++i)
        atomicAdd(&sScore[wave][rbase + i], tanhf(acc[i] + wb) * wv);
    }
    // softmax over the 16 neighbors (lanes 0..15)
    float sc = (lane < 16) ? sScore[wave][lane] : -3.0e38f;
    float mx = sc;
#pragma unroll
    for (int o = 8; o; o >>= 1) mx = fmaxf(mx, __shfl_xor(mx, o, 16));
    float ex = (lane < 16) ? __expf(sc - mx) : 0.f;
    float sm = ex;
#pragma unroll
    for (int o = 8; o; o >>= 1) sm += __shfl_xor(sm, o, 16);
    if (lane < 16) sScore[wave][lane] = ex / sm;

    // agg[c] = sum_k alpha[k] * msg[k][c]
    for (int c = lane; c < H2C; c += 32) {
      float s = 0.f;
#pragma unroll
      for (int k = 0; k < KNN; ++k) s += sScore[wave][k] * (float)sMsg[wave][k][c];
      agg [(size_t)m * H2C + c] = s;
      aggh[(size_t)m * H2C + c] = (_Float16)s;
    }
  }
}

// ---------------- phase 3: channel MLP chain (WMMA + TDM-staged B panels) ----------------
// One wave per block. A fragments live in VGPRs for a whole GEMM stage;
// B panels (16 cols x K) are TDM-staged into LDS, double buffered on TENSORcnt.
__global__ void __launch_bounds__(32)
mlp_head_kernel(const _Float16* __restrict__ aggh, const float* __restrict__ aggf,
                const _Float16* __restrict__ expT, const float* __restrict__ exp_b, const float* __restrict__ exp_bn,
                const float* __restrict__ ds1_dw, const float* __restrict__ ds1_bn1,
                const _Float16* __restrict__ pw1T, const float* __restrict__ ds1_pw_b,
                const float* __restrict__ ds1_bn2, const float* __restrict__ blk_bn1,
                const float* __restrict__ ds2_dw, const float* __restrict__ ds2_bn1,
                const _Float16* __restrict__ pw2T, const float* __restrict__ ds2_pw_b,
                const float* __restrict__ ds2_bn2, const float* __restrict__ blk_bn2,
                const _Float16* __restrict__ projT, const float* __restrict__ proj_b, const float* __restrict__ proj_bn,
                float* __restrict__ outp, int M) {
  __shared__ _Float16 sAct[2][16][E4C];     // 32 KB activation ping-pong
  __shared__ _Float16 sB[2][16 * E4C];      // 32 KB TDM-staged B panels

  int lane = threadIdx.x & 31;
  int col = lane & 15, rbase = (lane >> 4) << 3;
  int tiles = (M + 15) / 16;

  // one GEMM stage: C[16xNT*16] = A(16xKT*32, regs) x TDM-staged B panels
  auto stage = [&](const _Float16* Ain, const _Float16* BT, int Kdim,
                   auto KTc, int NT, auto&& epi) {
    constexpr int KT = decltype(KTc)::value;
    v16h afr[KT];
#pragma unroll
    for (int kt = 0; kt < KT; ++kt) afr[kt] = load_frag(Ain, E4C, kt * 32);
    unsigned nelem = 16u * (unsigned)Kdim;
    tdm_stage_1d(lds_addr_of(&sB[0][0]), BT, nelem);
    for (int nt = 0; nt < NT; ++nt) {
      if (nt + 1 < NT) {
        tdm_stage_1d(lds_addr_of(&sB[(nt + 1) & 1][0]),
                     BT + (size_t)(nt + 1) * 16 * Kdim, nelem);
        __builtin_amdgcn_s_wait_tensorcnt(1);   // panel nt landed
      } else {
        __builtin_amdgcn_s_wait_tensorcnt(0);   // last panel landed
      }
      v8f acc = {};
#pragma unroll
      for (int kt = 0; kt < KT; ++kt) {
        v16h b = load_frag(&sB[nt & 1][0], Kdim, kt * 32);
        acc = __builtin_amdgcn_wmma_f32_16x16x32_f16(false, afr[kt], false, b,
                                                     (short)0, acc, false, false);
      }
      epi(nt, acc);
    }
  };

  for (int t = blockIdx.x; t < tiles; t += gridDim.x) {
    int m0 = t * 16;
    // stage agg (16x128 f16) into buffer 0 (rows >= M zeroed)
    for (int j = lane; j < 16 * RNNC; j += 32) {
      int r = j >> 7, c = j & 127;
      int mm = m0 + r;
      sAct[0][r][c] = (mm < M) ? aggh[(size_t)mm * RNNC + c] : (_Float16)0.f;
    }

    // G1: agg @ exp_w (128->512); epi: +b, bn_exp, relu, ds1 dw, ds1_bn1, relu
    stage(&sAct[0][0][0], expT, RNNC, std::integral_constant<int, 4>{}, E4C / 16,
          [&](int nt, v8f acc) {
            int c = nt * 16 + col;
            float se, te; bn_coef(exp_bn, E4C, c, se, te);
            float s1, t1; bn_coef(ds1_bn1, E4C, c, s1, t1);
            float d0 = ds1_dw[c], d1 = ds1_dw[E4C + c], eb = exp_b[c];
#pragma unroll
            for (int i = 0; i < 8; ++i) {
              float z = se * (acc[i] + eb) + te;  z = z > 0.f ? z : 0.f;   // out0
              float w = s1 * (z * d0 + d1) + t1;  w = w > 0.f ? w : 0.f;   // ds1 pre-pw
              sAct[1][rbase + i][c] = (_Float16)w;
            }
          });
    // G2: @ ds1_pw (512->512); epi: +b, ds1_bn2, blk_bn1, ds2 dw, ds2_bn1, relu
    stage(&sAct[1][0][0], pw1T, E4C, std::integral_constant<int, 16>{}, E4C / 16,
          [&](int nt, v8f acc) {
            int c = nt * 16 + col;
            float s2, t2; bn_coef(ds1_bn2, E4C, c, s2, t2);
            float sb, tb; bn_coef(blk_bn1, E4C, c, sb, tb);
            float sn, tn; bn_coef(ds2_bn1, E4C, c, sn, tn);
            float d0 = ds2_dw[c], d1 = ds2_dw[E4C + c], pb = ds1_pw_b[c];
#pragma unroll
            for (int i = 0; i < 8; ++i) {
              float u = sb * (s2 * (acc[i] + pb) + t2) + tb;               // out1
              float w = sn * (u * d0 + d1) + tn;  w = w > 0.f ? w : 0.f;   // ds2 pre-pw
              sAct[0][rbase + i][c] = (_Float16)w;
            }
          });
    // G3: @ ds2_pw (512->512); epi: +b, ds2_bn2, blk_bn2
    stage(&sAct[0][0][0], pw2T, E4C, std::integral_constant<int, 16>{}, E4C / 16,
          [&](int nt, v8f acc) {
            int c = nt * 16 + col;
            float s2, t2; bn_coef(ds2_bn2, E4C, c, s2, t2);
            float sb, tb; bn_coef(blk_bn2, E4C, c, sb, tb);
            float pb = ds2_pw_b[c];
#pragma unroll
            for (int i = 0; i < 8; ++i)
              sAct[1][rbase + i][c] = (_Float16)(sb * (s2 * (acc[i] + pb) + t2) + tb);
          });
    // G4: @ proj_w (512->128); epi: +b, proj_bn, +agg residual, relu -> d_out
    stage(&sAct[1][0][0], projT, E4C, std::integral_constant<int, 16>{}, RNNC / 16,
          [&](int nt, v8f acc) {
            int c = nt * 16 + col;
            float sp, tp; bn_coef(proj_bn, RNNC, c, sp, tp);
            float pb = proj_b[c];
#pragma unroll
            for (int i = 0; i < 8; ++i) {
              int mm = m0 + rbase + i;
              if (mm < M) {
                float r = sp * (acc[i] + pb) + tp + aggf[(size_t)mm * RNNC + c];
                outp[(size_t)mm * RNNC + c] = r > 0.f ? r : 0.f;
              }
            }
          });
  }
}

// ---------------- tail outputs: pos[idx], batch[idx], refl_g[idx] ----------------
__global__ void tail_kernel(const float* __restrict__ pos, const int* __restrict__ batch,
                            const float* __restrict__ pos4, const int* __restrict__ idx,
                            float* __restrict__ out, int M) {
  int m = blockIdx.x * blockDim.x + threadIdx.x;
  if (m >= M) return;
  int p = idx[m];
  float* posO = out + (size_t)M * RNNC;
  posO[m * 3 + 0] = pos[p * 3 + 0];
  posO[m * 3 + 1] = pos[p * 3 + 1];
  posO[m * 3 + 2] = pos[p * 3 + 2];
  float* bO = posO + (size_t)M * 3;
  bO[m] = (float)batch[p];
  float* rO = bO + M;
  rO[m] = pos4[p * 4 + 3];
}

// ---------------- launch ----------------
extern "C" void kernel_launch(void* const* d_in, const int* in_sizes, int n_in,
                              void* d_out, int out_size, void* d_ws, size_t ws_size,
                              hipStream_t stream) {
  const float* x      = (const float*)d_in[0];
  const float* pos    = (const float*)d_in[1];
  const float* refl   = (const float*)d_in[2];
  const float* gum    = (const float*)d_in[3];
  const float* gw1    = (const float*)d_in[4];
  const float* gb1    = (const float*)d_in[5];
  const float* gw2    = (const float*)d_in[6];
  const float* gb2    = (const float*)d_in[7];
  const float* gwg    = (const float*)d_in[8];
  const float* gbg    = (const float*)d_in[9];
  const float* loc_w1 = (const float*)d_in[10];
  const float* loc_b1 = (const float*)d_in[11];
  const float* loc_w2 = (const float*)d_in[12];
  const float* loc_b2 = (const float*)d_in[13];
  const float* attn_w1= (const float*)d_in[14];
  const float* attn_b1= (const float*)d_in[15];
  const float* attn_w2= (const float*)d_in[16];
  const float* exp_w  = (const float*)d_in[17];
  const float* exp_b  = (const float*)d_in[18];
  const float* exp_bn = (const float*)d_in[19];
  const float* ds1_dw = (const float*)d_in[20];
  const float* ds1_bn1= (const float*)d_in[21];
  const float* ds1_pw = (const float*)d_in[22];
  const float* ds1_pwb= (const float*)d_in[23];
  const float* ds1_bn2= (const float*)d_in[24];
  const float* blk_bn1= (const float*)d_in[25];
  const float* ds2_dw = (const float*)d_in[26];
  const float* ds2_bn1= (const float*)d_in[27];
  const float* ds2_pw = (const float*)d_in[28];
  const float* ds2_pwb= (const float*)d_in[29];
  const float* ds2_bn2= (const float*)d_in[30];
  const float* blk_bn2= (const float*)d_in[31];
  const float* proj_w = (const float*)d_in[32];
  const float* proj_b = (const float*)d_in[33];
  const float* proj_bn= (const float*)d_in[34];
  const int*   batch  = (const int*)d_in[35];
  const int*   idx    = (const int*)d_in[36];
  const int*   eidx   = (const int*)d_in[37];

  int N = in_sizes[2];
  int M = in_sizes[36];

  size_t off = 0;
  auto alloc = [&](size_t bytes) -> char* {
    off = (off + 255) & ~(size_t)255;
    char* p = (char*)d_ws + off;
    off += bytes;
    return p;
  };
  unsigned*  pooled = (unsigned*)alloc((size_t)NBATCH * H1G * 4);
  float*     gsel   = (float*)   alloc((size_t)NBATCH * 4);
  float*     pos4   = (float*)   alloc((size_t)N * 4 * 4);
  float*     agg    = (float*)   alloc((size_t)M * H2C * 4);
  _Float16*  aggh   = (_Float16*)alloc((size_t)M * H2C * 2);
  _Float16*  w1T    = (_Float16*)alloc((size_t)H2C * KP1 * 2);
  _Float16*  w2T    = (_Float16*)alloc((size_t)H2C * H2C * 2);
  _Float16*  aT     = (_Float16*)alloc((size_t)ATTN_C * H2C * 2);
  _Float16*  expT   = (_Float16*)alloc((size_t)E4C * RNNC * 2);
  _Float16*  pw1T   = (_Float16*)alloc((size_t)E4C * E4C * 2);
  _Float16*  pw2T   = (_Float16*)alloc((size_t)E4C * E4C * 2);
  _Float16*  projT  = (_Float16*)alloc((size_t)RNNC * E4C * 2);
  (void)ws_size; (void)n_in; (void)out_size;

  // weight conversion (f32 [K][N] -> f16 [N][Kpad])
  auto cvt = [&](const float* s, _Float16* d, int K, int Nn, int Kp) {
    int tot = Nn * Kp;
    hipLaunchKernelGGL(cvtT_kernel, dim3((tot + 255) / 256), dim3(256), 0, stream, s, d, K, Nn, Kp);
  };
  hipLaunchKernelGGL(zero_u32_kernel, dim3((NBATCH * H1G + 255) / 256), dim3(256), 0, stream,
                     pooled, NBATCH * H1G);
  cvt(loc_w1, w1T, H1G, H2C, KP1);
  cvt(loc_w2, w2T, H2C, H2C, H2C);
  cvt(attn_w1, aT, H2C, ATTN_C, H2C);
  cvt(exp_w, expT, RNNC, E4C, RNNC);
  cvt(ds1_pw, pw1T, E4C, E4C, E4C);
  cvt(ds2_pw, pw2T, E4C, E4C, E4C);
  cvt(proj_w, projT, E4C, RNNC, E4C);

  hipLaunchKernelGGL(gate_mlp_kernel, dim3((N + 127) / 128), dim3(128), 0, stream,
                     pos, refl, batch, gw1, gb1, gw2, gb2, pooled, N);
  hipLaunchKernelGGL(gate_softmax_kernel, dim3(1), dim3(32), 0, stream,
                     pooled, gwg, gbg, gum, gsel);
  hipLaunchKernelGGL(pos4_kernel, dim3((N + 255) / 256), dim3(256), 0, stream,
                     pos, refl, batch, gsel, pos4, N);

  int tiles2 = (M + P2_WAVES - 1) / P2_WAVES;
  int blk2 = tiles2 < 2048 ? tiles2 : 2048;
  hipLaunchKernelGGL(edge_attn_kernel, dim3(blk2), dim3(P2_WAVES * 32), 0, stream,
                     x, pos4, idx, eidx, w1T, loc_b1, w2T, loc_b2, aT, attn_b1, attn_w2,
                     agg, aggh, M);

  int tiles3 = (M + 15) / 16;
  int blk3 = tiles3 < 2048 ? tiles3 : 2048;
  hipLaunchKernelGGL(mlp_head_kernel, dim3(blk3), dim3(32), 0, stream,
                     aggh, agg, expT, exp_b, exp_bn,
                     ds1_dw, ds1_bn1, pw1T, ds1_pwb, ds1_bn2, blk_bn1,
                     ds2_dw, ds2_bn1, pw2T, ds2_pwb, ds2_bn2, blk_bn2,
                     projT, proj_b, proj_bn, (float*)d_out, M);

  hipLaunchKernelGGL(tail_kernel, dim3((M + 255) / 256), dim3(256), 0, stream,
                     pos, batch, pos4, idx, (float*)d_out, M);
}
